// CROMA_61091614818438
// MI455X (gfx1250) — matrix operand order repaired and checked by reading, not compiled, not run on hardware
//
#include <hip/hip_runtime.h>
#include <math.h>

// ---------------- problem dims ----------------
#define BB 64
#define NN 196
#define DD 768
#define HH 16
#define DHD 48
#define LL 4
#define FFD 3072
#define MM (BB*NN)      // 12544 = 98*128
#define NP 224          // padded seq (14*16 = 7*32)
#define BH (BB*HH)      // 1024
#define QK_SCALE 0.14433756729740643f  // 48^-0.5

typedef __attribute__((ext_vector_type(16))) __bf16 v16bf;
typedef __attribute__((ext_vector_type(8)))  float  v8f;

// Build a WMMA 16-element bf16 fragment from two 16-byte LDS/global chunks.
__device__ inline v16bf ld_frag(const __bf16* p0, const __bf16* p1) {
  union { v16bf v; uint4 u[2]; } x;
  x.u[0] = *(const uint4*)p0;
  x.u[1] = *(const uint4*)p1;
  return x.v;
}

__device__ inline v8f wmma_bf16(v16bf a, v16bf b, v8f c) {
  return __builtin_amdgcn_wmma_f32_16x16x32_bf16(false, a, false, b, (short)0, c,
                                                 false, false);
}

__device__ inline float gelu_f(float x) {
  return 0.5f * x * (1.0f + erff(x * 0.70710678118654752f));
}

// ---------------- LayerNorm ----------------
// one block (256 threads) per row of 768; OUTF32 ? fp32 out : bf16 out
template<bool OUTF32>
__global__ __launch_bounds__(256)
void ln_kernel(const float* __restrict__ in, const float* __restrict__ g,
               const float* __restrict__ b, void* __restrict__ outp)
{
  __shared__ float red[256];
  const int row = blockIdx.x, tid = threadIdx.x;
  const float* x = in + (size_t)row * DD;
  float v0 = x[tid], v1 = x[tid + 256], v2 = x[tid + 512];
  red[tid] = v0 + v1 + v2;
  __syncthreads();
  for (int o = 128; o > 0; o >>= 1) { if (tid < o) red[tid] += red[tid + o]; __syncthreads(); }
  float mu = red[0] * (1.0f / DD);
  __syncthreads();
  float d0 = v0 - mu, d1 = v1 - mu, d2 = v2 - mu;
  red[tid] = d0 * d0 + d1 * d1 + d2 * d2;
  __syncthreads();
  for (int o = 128; o > 0; o >>= 1) { if (tid < o) red[tid] += red[tid + o]; __syncthreads(); }
  float rs = rsqrtf(red[0] * (1.0f / DD) + 1e-5f);
  #pragma unroll
  for (int i = 0; i < 3; ++i) {
    int c = tid + i * 256;
    float vv = (i == 0 ? v0 : (i == 1 ? v1 : v2));
    float o = (vv - mu) * rs * g[c] + b[c];
    if (OUTF32) ((float*)outp)[(size_t)row * DD + c] = o;
    else        ((__bf16*)outp)[(size_t)row * DD + c] = (__bf16)o;
  }
}

// ---------------- WMMA GEMM ----------------
// C[M x N] = A(bf16)[M x K] * W(f32)[K x N]; block tile 128x128, 8 waves (2x4),
// each wave 64x32 = 8 C-tiles of 16x16. Weights converted to bf16 + transposed
// during LDS staging so B fragments are contiguous 32B per lane.
// EPI: 0 = bf16 store, 1 = gelu -> bf16 store, 2 = bias + residual RMW (fp32)
template<int EPI>
__global__ __launch_bounds__(256)
void gemm_kernel(const __bf16* __restrict__ A, const float* __restrict__ W,
                 int ldw, const float* __restrict__ bias,
                 float* __restrict__ resid, __bf16* __restrict__ outb,
                 int ldo, int K)
{
  constexpr int LDS_STRIDE = 40;  // bf16 elems per LDS row (pad for banks)
  __shared__ __bf16 As[128 * LDS_STRIDE];
  __shared__ __bf16 Ws[128 * LDS_STRIDE];

  const int tid  = threadIdx.x;
  const int lane = tid & 31, wave = tid >> 5;
  const int wrow = wave >> 2, wcol = wave & 3;
  const int m0 = blockIdx.x * 128;
  const int n0 = blockIdx.y * 128;

  const v8f vz = {0.f, 0.f, 0.f, 0.f, 0.f, 0.f, 0.f, 0.f};
  v8f acc[4][2];
  #pragma unroll
  for (int mt = 0; mt < 4; ++mt) { acc[mt][0] = vz; acc[mt][1] = vz; }

  const int arow = tid >> 1, ahalf = tid & 1;     // A staging: 128 rows x 2 halves
  const int wk = tid >> 7, wn = tid & 127;        // W staging: 2 k-rows x 128 cols
  const int r   = lane & 15;
  const int alo = (lane < 16) ? 0 : 16;
  const int blo = (lane < 16) ? 0 : 16;

  for (int k0 = 0; k0 < K; k0 += 32) {
    // stage A tile [128 x 32] bf16 (32B per thread)
    {
      const uint4* src = (const uint4*)(A + (size_t)(m0 + arow) * K + k0 + ahalf * 16);
      uint4* dst = (uint4*)(As + arow * LDS_STRIDE + ahalf * 16);
      dst[0] = src[0];
      dst[1] = src[1];
      if (k0 + 32 < K)
        __builtin_prefetch(A + (size_t)(m0 + arow) * K + (k0 + 32) + ahalf * 16, 0, 0);
    }
    // stage W tile: convert f32->bf16 and transpose to [n][k]
    #pragma unroll
    for (int p = 0; p < 16; ++p) {
      int k = p * 2 + wk;
      float w = W[(size_t)(k0 + k) * ldw + n0 + wn];
      Ws[wn * LDS_STRIDE + k] = (__bf16)w;
    }
    __syncthreads();

    v16bf bfr[2];
    #pragma unroll
    for (int nt = 0; nt < 2; ++nt) {
      const __bf16* pb = Ws + (size_t)(wcol * 32 + nt * 16 + r) * LDS_STRIDE + blo;
      bfr[nt] = ld_frag(pb, pb + 8);
    }
    #pragma unroll
    for (int mt = 0; mt < 4; ++mt) {
      const __bf16* pa = As + (size_t)(wrow * 64 + mt * 16 + r) * LDS_STRIDE + alo;
      v16bf af = ld_frag(pa, pa + 8);
      acc[mt][0] = wmma_bf16(af, bfr[0], acc[mt][0]);
      acc[mt][1] = wmma_bf16(af, bfr[1], acc[mt][1]);
    }
    __syncthreads();
  }

  // epilogue: C layout -> elem e: row = (lane<16 ? e : 8+e), col = lane%16
  const int rofs = (lane < 16) ? 0 : 8;
  #pragma unroll
  for (int mt = 0; mt < 4; ++mt) {
    #pragma unroll
    for (int nt = 0; nt < 2; ++nt) {
      int gn = n0 + wcol * 32 + nt * 16 + r;
      float bv = bias ? bias[gn] : 0.f;
      #pragma unroll
      for (int e = 0; e < 8; ++e) {
        int gm = m0 + wrow * 64 + mt * 16 + rofs + e;
        float v = acc[mt][nt][e] + bv;
        if constexpr (EPI == 1) v = gelu_f(v);
        if constexpr (EPI == 2) {
          float* p = resid + (size_t)gm * DD + gn;
          *p = *p + v;
        } else {
          outb[(size_t)gm * ldo + gn] = (__bf16)v;
        }
      }
    }
  }
}

// ---------------- qkv -> padded attention layouts ----------------
// qp, kp : [BH][NP][64] bf16 (zero padded rows/cols)
// vT     : [BH][64][NP] bf16 (transposed so PV B-fragments are contiguous)
__global__ __launch_bounds__(256)
void scatter_qkv(const __bf16* __restrict__ qkv,  // [MM][2304]
                 __bf16* __restrict__ qp, __bf16* __restrict__ kp,
                 __bf16* __restrict__ vT)
{
  int idx = blockIdx.x * 256 + threadIdx.x;   // over BH*NP*64
  int dh = idx & 63;
  int n  = (idx >> 6) % NP;
  int bh = idx / (64 * NP);
  int b = bh >> 4, h = bh & 15;
  __bf16 q = (__bf16)0.f, k = q, v = q;
  if (n < NN && dh < DHD) {
    size_t src = (size_t)(b * NN + n) * (3 * DD) + h * DHD + dh;
    q = qkv[src];
    k = qkv[src + DD];
    v = qkv[src + 2 * DD];
  }
  qp[((size_t)bh * NP + n) * 64 + dh] = q;
  kp[((size_t)bh * NP + n) * 64 + dh] = k;
  vT[((size_t)bh * 64 + dh) * NP + n] = v;
}

// ---------------- fused attention ----------------
// grid (7 row-blocks of 32, BH).
// 1) stage q rows + ALiBi block (into S) behind one barrier
// 2) S += (q k^T) * scale via WMMA -- branch-free LDS RMW epilogue
// 3) softmax over the 196 real columns only; P cols >=196 zeroed
// 4) O = P V via WMMA (V pre-transposed so B-fragments are contiguous)
// Padded rows >=196 produce garbage the merge kernel never reads.
__global__ __launch_bounds__(256)
void attn_kernel(const __bf16* __restrict__ qp, const __bf16* __restrict__ kp,
                 const __bf16* __restrict__ vT, const float* __restrict__ alibi,
                 __bf16* __restrict__ opad)
{
  __shared__ __bf16 qs[32 * 72];       // 32 q-rows x 64 (stride 72)
  __shared__ float  S[32 * 224];       // alibi then scores
  __shared__ __bf16 P[32 * 232];       // probs bf16 (stride 232)

  const int tid = threadIdx.x, lane = tid & 31, wave = tid >> 5;
  const int bh = blockIdx.y;
  const int h  = bh & 15;
  const int rowBase = blockIdx.x * 32;
  const __bf16* qb = qp + (size_t)bh * NP * 64;
  const __bf16* kb = kp + (size_t)bh * NP * 64;
  const __bf16* vb = vT + (size_t)bh * 64 * NP;

  {  // stage 32 q rows (one uint4 per thread)
    int row = tid >> 3, seg = tid & 7;
    *(uint4*)(qs + row * 72 + seg * 8) =
        *(const uint4*)(qb + (size_t)(rowBase + row) * 64 + seg * 8);
  }
  {  // stage ALiBi block into S (cols < 196; cols >=196 never read)
    int row = tid >> 3, sub = tid & 7;
    int qi = rowBase + row;
    int qc = qi < NN ? qi : (NN - 1);
    const float* arow = alibi + ((size_t)h * NN + qc) * NN;
    for (int c = sub; c < NN; c += 8) S[row * 224 + c] = arow[c];
  }
  __syncthreads();

  const int r   = lane & 15;
  const int alo = (lane < 16) ? 0 : 16;
  const int blo = (lane < 16) ? 0 : 16;
  const v8f vz = {0.f, 0.f, 0.f, 0.f, 0.f, 0.f, 0.f, 0.f};

  // preload q fragments once: [mt][kk]
  v16bf qfrag[2][2];
  #pragma unroll
  for (int mt = 0; mt < 2; ++mt) {
    #pragma unroll
    for (int kk = 0; kk < 2; ++kk) {
      const __bf16* pa = qs + (size_t)(mt * 16 + r) * 72 + kk * 32 + alo;
      qfrag[mt][kk] = ld_frag(pa, pa + 8);
    }
  }

  // ---- scores: waves own key tiles jt (14 tiles of 16) ----
  for (int jt = wave; jt < 14; jt += 8) {
    int scol = jt * 16 + r;
    #pragma unroll
    for (int mt = 0; mt < 2; ++mt) {
      v8f acc = vz;
      #pragma unroll
      for (int kk = 0; kk < 2; ++kk) {   // DH padded to 64 = 2 x 32
        const __bf16* pb = kb + (size_t)(jt * 16 + r) * 64 + kk * 32 + blo;
        v16bf bf = ld_frag(pb, pb + 8);
        acc = wmma_bf16(qfrag[mt][kk], bf, acc);
      }
      // branch-free: S already holds alibi; accumulate scaled scores
      #pragma unroll
      for (int e = 0; e < 8; ++e) {
        int srow = mt * 16 + ((lane < 16) ? e : 8 + e);
        S[srow * 224 + scol] += acc[e] * QK_SCALE;
      }
    }
  }
  __syncthreads();

  // ---- softmax over 196 real columns; 8 threads per row ----
  {
    int row = tid >> 3, sub = tid & 7;
    float m = -1e30f;
    for (int c = sub; c < NN; c += 8) m = fmaxf(m, S[row * 224 + c]);
    for (int o = 4; o >= 1; o >>= 1) m = fmaxf(m, __shfl_xor(m, o, 8));
    float s = 0.f;
    for (int c = sub; c < NN; c += 8) {
      float e = __expf(S[row * 224 + c] - m);
      s += e;
      P[row * 232 + c] = (__bf16)e;
    }
    for (int o = 4; o >= 1; o >>= 1) s += __shfl_xor(s, o, 8);
    float rinv = 1.f / s;
    for (int c = sub; c < NN; c += 8)
      P[row * 232 + c] = (__bf16)((float)P[row * 232 + c] * rinv);
    for (int c = NN + sub; c < 224; c += 8)       // zero padded prob columns
      P[row * 232 + c] = (__bf16)0.f;
  }
  __syncthreads();

  // ---- O = P * V : 8 waves = 2 row-tiles x 4 col-tiles ----
  {
    int mt = wave >> 2, nt = wave & 3;
    v8f acc = vz;
    #pragma unroll
    for (int ks = 0; ks < 7; ++ks) {   // 224 = 7 x 32
      const __bf16* pa = P + (size_t)(mt * 16 + r) * 232 + ks * 32 + alo;
      v16bf af = ld_frag(pa, pa + 8);
      const __bf16* pb = vb + (size_t)(nt * 16 + r) * NP + ks * 32 + blo;
      v16bf bf = ld_frag(pb, pb + 8);
      acc = wmma_bf16(af, bf, acc);
    }
    __bf16* ob = opad + (size_t)bh * NP * 64;
    #pragma unroll
    for (int e = 0; e < 8; ++e) {
      int orow = rowBase + mt * 16 + ((lane < 16) ? e : 8 + e);
      int ocol = nt * 16 + r;
      ob[(size_t)orow * 64 + ocol] = (__bf16)acc[e];
    }
  }
}

// [BH][NP][64] padded -> [MM][768] merged heads
__global__ __launch_bounds__(256)
void merge_kernel(const __bf16* __restrict__ opad, __bf16* __restrict__ abf)
{
  int idx = blockIdx.x * 256 + threadIdx.x;   // over MM*DD
  int c = idx % DD, m = idx / DD;
  int b = m / NN, n = m % NN;
  int h = c / DHD, dh = c % DHD;
  abf[idx] = opad[(((size_t)(b * HH + h)) * NP + n) * 64 + dh];
}

// ---------------- host orchestration ----------------
extern "C" void kernel_launch(void* const* d_in, const int* in_sizes, int n_in,
                              void* d_out, int out_size, void* d_ws, size_t ws_size,
                              hipStream_t stream)
{
  (void)in_sizes; (void)n_in; (void)out_size; (void)ws_size;
  const float* x_in     = (const float*)d_in[0];
  const float* ctx_in   = (const float*)d_in[1];
  const float* alibi    = (const float*)d_in[2];
  const float* sa_ln_g  = (const float*)d_in[3];
  const float* sa_ln_b  = (const float*)d_in[4];
  const float* sa_qkv_w = (const float*)d_in[5];
  const float* sa_out_w = (const float*)d_in[6];
  const float* sa_out_b = (const float*)d_in[7];
  const float* ca_ln_g  = (const float*)d_in[8];
  const float* ca_ln_b  = (const float*)d_in[9];
  const float* ca_q_w   = (const float*)d_in[10];
  const float* ca_k_w   = (const float*)d_in[11];
  const float* ca_v_w   = (const float*)d_in[12];
  const float* ca_out_w = (const float*)d_in[13];
  const float* ca_out_b = (const float*)d_in[14];
  const float* ff_ln_g  = (const float*)d_in[15];
  const float* ff_ln_b  = (const float*)d_in[16];
  const float* ff_w1    = (const float*)d_in[17];
  const float* ff_b1    = (const float*)d_in[18];
  const float* ff_w2    = (const float*)d_in[19];
  const float* ff_b2    = (const float*)d_in[20];
  const float* norm_g   = (const float*)d_in[21];
  const float* norm_b   = (const float*)d_in[22];

  float* xf = (float*)d_out;  // residual stream lives in d_out (fp32)

  size_t off = 0;
  auto alloc = [&](size_t bytes) -> char* {
    char* p = (char*)d_ws + off;
    off += (bytes + 255) & ~(size_t)255;
    return p;
  };
  __bf16* hbf  = (__bf16*)alloc((size_t)MM * DD * 2);        // LN(x) bf16
  __bf16* hcbf = (__bf16*)alloc((size_t)MM * DD * 2);        // LN(context) bf16
  __bf16* big  = (__bf16*)alloc((size_t)MM * FFD * 2);       // qkv / ffn hidden (aliased, disjoint lifetimes)
  __bf16* qkvbuf = big;
  __bf16* ffh    = big;
  __bf16* qp   = (__bf16*)alloc((size_t)BH * NP * 64 * 2);
  __bf16* kp   = (__bf16*)alloc((size_t)BH * NP * 64 * 2);
  __bf16* vT   = (__bf16*)alloc((size_t)BH * 64 * NP * 2);
  __bf16* opad = (__bf16*)alloc((size_t)BH * NP * 64 * 2);
  __bf16* abf  = (__bf16*)alloc((size_t)MM * DD * 2);

  hipMemcpyAsync(xf, x_in, (size_t)MM * DD * 4, hipMemcpyDeviceToDevice, stream);

  const dim3 blk(256);
  const dim3 g_dd(MM / 128, DD / 128);
  const dim3 g_qkv(MM / 128, (3 * DD) / 128);
  const dim3 g_ff(MM / 128, FFD / 128);
  const int scatter_blocks = (BH * NP * 64) / 256;
  const int merge_blocks   = (MM * DD) / 256;

  for (int l = 0; l < LL; ++l) {
    // ---- self attention ----
    ln_kernel<false><<<MM, blk, 0, stream>>>(xf, sa_ln_g + l * DD, sa_ln_b + l * DD, hbf);
    gemm_kernel<0><<<g_qkv, blk, 0, stream>>>(hbf, sa_qkv_w + (size_t)l * DD * 3 * DD,
                                              3 * DD, nullptr, nullptr, qkvbuf, 3 * DD, DD);
    scatter_qkv<<<scatter_blocks, blk, 0, stream>>>(qkvbuf, qp, kp, vT);
    attn_kernel<<<dim3(7, BH), blk, 0, stream>>>(qp, kp, vT, alibi, opad);
    merge_kernel<<<merge_blocks, blk, 0, stream>>>(opad, abf);
    gemm_kernel<2><<<g_dd, blk, 0, stream>>>(abf, sa_out_w + (size_t)l * DD * DD, DD,
                                             sa_out_b + l * DD, xf, nullptr, 0, DD);
    // ---- cross attention (shared LN params for x and context) ----
    ln_kernel<false><<<MM, blk, 0, stream>>>(xf, ca_ln_g + l * DD, ca_ln_b + l * DD, hbf);
    ln_kernel<false><<<MM, blk, 0, stream>>>(ctx_in, ca_ln_g + l * DD, ca_ln_b + l * DD, hcbf);
    gemm_kernel<0><<<g_dd, blk, 0, stream>>>(hbf,  ca_q_w + (size_t)l * DD * DD, DD,
                                             nullptr, nullptr, qkvbuf, 3 * DD, DD);
    gemm_kernel<0><<<g_dd, blk, 0, stream>>>(hcbf, ca_k_w + (size_t)l * DD * DD, DD,
                                             nullptr, nullptr, qkvbuf + DD, 3 * DD, DD);
    gemm_kernel<0><<<g_dd, blk, 0, stream>>>(hcbf, ca_v_w + (size_t)l * DD * DD, DD,
                                             nullptr, nullptr, qkvbuf + 2 * DD, 3 * DD, DD);
    scatter_qkv<<<scatter_blocks, blk, 0, stream>>>(qkvbuf, qp, kp, vT);
    attn_kernel<<<dim3(7, BH), blk, 0, stream>>>(qp, kp, vT, alibi, opad);
    merge_kernel<<<merge_blocks, blk, 0, stream>>>(opad, abf);
    gemm_kernel<2><<<g_dd, blk, 0, stream>>>(abf, ca_out_w + (size_t)l * DD * DD, DD,
                                             ca_out_b + l * DD, xf, nullptr, 0, DD);
    // ---- FFN ----
    ln_kernel<false><<<MM, blk, 0, stream>>>(xf, ff_ln_g + l * DD, ff_ln_b + l * DD, hbf);
    gemm_kernel<1><<<g_ff, blk, 0, stream>>>(hbf, ff_w1 + (size_t)l * DD * FFD, FFD,
                                             ff_b1 + l * FFD, nullptr, ffh, FFD, DD);
    gemm_kernel<2><<<g_dd, blk, 0, stream>>>(ffh, ff_w2 + (size_t)l * FFD * DD, DD,
                                             ff_b2 + l * DD, xf, nullptr, 0, FFD);
  }
  // final LN in place (reads into registers before writing -> safe)
  ln_kernel<true><<<MM, blk, 0, stream>>>(xf, norm_g, norm_b, xf);
}
